// HouseholderLinear_60017872994386
// MI455X (gfx1250) — matrix operand order repaired and checked by reading, not compiled, not run on hardware
//
#include <hip/hip_runtime.h>

// ---------------------------------------------------------------------------
// HouseholderLinear on MI455X (gfx1250, wave32, WMMA):
//   out = (x - 2 (x.v) v) @ W + bias
// Compute-bound GEMM (AI ~820 flop/byte) on the bf16 WMMA pipe with f32
// accumulate. Householder fused in f32 before the bf16 down-convert.
// Block tile 128x256, 8 waves (2x4), 64x64 per wave = 16 wmma / K-step.
// Software-pipelined global loads; packed bf16 LDS staging.
// ---------------------------------------------------------------------------

typedef __bf16 v16bf __attribute__((ext_vector_type(16)));
typedef float  v8f   __attribute__((ext_vector_type(8)));

union FragU { v16bf v; uint4 u[2]; };

#define N_IN   4096
#define N_OUT  4096
#define BATCH  8192

#define BM 128
#define BN 256
#define BK 32
#define LDA 40   // ushort stride, A tile [m][k]   (32 + 8 pad, 16B-aligned rows)
#define LDB 40   // ushort stride, B tile [n][k]   (transposed)

// pack two f32 -> packed bf16 pair (RNE), lo in [15:0], hi in [31:16]
__device__ __forceinline__ unsigned int pk2bf(float lo, float hi) {
  unsigned int a = __float_as_uint(lo);
  unsigned int b = __float_as_uint(hi);
  a = (a + 0x7FFFu + ((a >> 16) & 1u)) >> 16;
  b = (b + 0x7FFFu + ((b >> 16) & 1u)) & 0xFFFF0000u;
  return a | b;
}

// ---------------------------------------------------------------------------
// Kernel 1: vn = vs[time] / ||vs[time]||   (single block)
// ---------------------------------------------------------------------------
__global__ __launch_bounds__(256)
void k_norm(const float* __restrict__ vs, const int* __restrict__ timep,
            float* __restrict__ vn) {
  __shared__ float red[256];
  const int tid = threadIdx.x;
  const int t = timep[0];
  const float* v = vs + (size_t)t * N_IN;
  float s = 0.f;
  for (int i = tid; i < N_IN; i += 256) { float a = v[i]; s += a * a; }
  red[tid] = s;
  __syncthreads();
  for (int o = 128; o > 0; o >>= 1) {
    if (tid < o) red[tid] += red[tid + o];
    __syncthreads();
  }
  const float r = rsqrtf(red[0]);
  for (int i = tid; i < N_IN; i += 256) vn[i] = v[i] * r;
}

// ---------------------------------------------------------------------------
// Kernel 2: xv[b] = dot(x[b,:], vn)   (one wave32 per row, 8 rows / block)
// ---------------------------------------------------------------------------
__global__ __launch_bounds__(256)
void k_xv(const float* __restrict__ x, const float* __restrict__ vn,
          float* __restrict__ xv) {
  const int lane = threadIdx.x & 31;
  const int wv   = threadIdx.x >> 5;
  const size_t row = (size_t)blockIdx.x * 8 + wv;
  const float* xr = x + row * N_IN;
  float s = 0.f;
  for (int i = lane * 4; i < N_IN; i += 32 * 4) {
    float4 a = *(const float4*)(xr + i);
    float4 b = *(const float4*)(vn + i);
    s += a.x * b.x + a.y * b.y + a.z * b.z + a.w * b.w;
  }
  for (int o = 16; o > 0; o >>= 1) s += __shfl_xor(s, o, 32);
  if (lane == 0) xv[row] = s;
}

// ---------------------------------------------------------------------------
// Kernel 3: out = M @ W + bias, M[m][k] = x[m][k] - 2*xv[m]*vn[k] (bf16 WMMA)
// ---------------------------------------------------------------------------
__global__ __launch_bounds__(256)
void k_gemm(const float* __restrict__ x, const float* __restrict__ w,
            const float* __restrict__ bias, const float* __restrict__ vn,
            const float* __restrict__ xv, float* __restrict__ out) {
  __shared__ __align__(16) unsigned short sA[BM * LDA];  // [m][k]
  __shared__ __align__(16) unsigned short sB[BN * LDB];  // [n][k] (transposed)

  const int tid  = threadIdx.x;
  const int lane = tid & 31;
  const int wave = tid >> 5;
  const int wm = wave >> 2;            // 0..1 : 64-row strip
  const int wn = wave & 3;             // 0..3 : 64-col strip
  const int bm = blockIdx.y * BM;
  const int bn = blockIdx.x * BN;

  v8f acc[4][4] = {};

  // ---- A stager: 4 passes of 32 rows, float4 along k ----
  const int lrowA = tid >> 3;          // 0..31
  const int lcolA = (tid & 7) * 4;     // 0,4,...,28
  float hs[4];
  const float* xrow[4];
#pragma unroll
  for (int p = 0; p < 4; ++p) {
    const int grow = bm + lrowA + p * 32;
    hs[p] = -2.0f * xv[grow];
    xrow[p] = x + (size_t)grow * N_IN;
  }

  // ---- B stager: two consecutive k-rows x 16 n per thread ----
  const int kr0 = (tid >> 4) * 2;      // 0,2,...,30
  const int nb  = (tid & 15) * 16;     // 0..240
  const float* wrow0 = w + (size_t)kr0 * N_OUT + bn + nb;
  const float* wrow1 = wrow0 + N_OUT;

  // ---- fragment geometry (ISA 16-bit A/B layouts) ----
  const int ar = lane & 15;
  const int ak = (lane < 16) ? 0 : 8;   // A K-chunk select
  const int bk = (lane < 16) ? 0 : 8;   // B K-chunk select

  // ---- software pipeline: prologue loads for k0 = 0 ----
  float4 xa[4], wb0[4], wb1[4], vv;
  vv = *(const float4*)(vn + lcolA);
#pragma unroll
  for (int p = 0; p < 4; ++p) xa[p] = *(const float4*)(xrow[p] + lcolA);
#pragma unroll
  for (int q = 0; q < 4; ++q) {
    wb0[q] = *(const float4*)(wrow0 + q * 4);
    wb1[q] = *(const float4*)(wrow1 + q * 4);
  }

  for (int k0 = 0; k0 < N_IN; k0 += BK) {
    // ---- stage current K-step into LDS (packed bf16) ----
#pragma unroll
    for (int p = 0; p < 4; ++p) {
      const float m0 = fmaf(hs[p], vv.x, xa[p].x);
      const float m1 = fmaf(hs[p], vv.y, xa[p].y);
      const float m2 = fmaf(hs[p], vv.z, xa[p].z);
      const float m3 = fmaf(hs[p], vv.w, xa[p].w);
      uint2 pk;
      pk.x = pk2bf(m0, m1);
      pk.y = pk2bf(m2, m3);
      *(uint2*)&sA[(lrowA + p * 32) * LDA + lcolA] = pk;   // ds_store_b64
    }
#pragma unroll
    for (int q = 0; q < 4; ++q) {
      const int n = nb + q * 4;
      *(unsigned int*)&sB[(n + 0) * LDB + kr0] = pk2bf(wb0[q].x, wb1[q].x);
      *(unsigned int*)&sB[(n + 1) * LDB + kr0] = pk2bf(wb0[q].y, wb1[q].y);
      *(unsigned int*)&sB[(n + 2) * LDB + kr0] = pk2bf(wb0[q].z, wb1[q].z);
      *(unsigned int*)&sB[(n + 3) * LDB + kr0] = pk2bf(wb0[q].w, wb1[q].w);
    }
    __syncthreads();

    // ---- issue next K-step's global loads (hide under the WMMAs) ----
    const int kn = (k0 + BK < N_IN) ? (k0 + BK) : 0;
    vv = *(const float4*)(vn + kn + lcolA);
#pragma unroll
    for (int p = 0; p < 4; ++p) xa[p] = *(const float4*)(xrow[p] + kn + lcolA);
    {
      const float* w0 = wrow0 + (size_t)kn * N_OUT;
      const float* w1 = wrow1 + (size_t)kn * N_OUT;
#pragma unroll
      for (int q = 0; q < 4; ++q) {
        wb0[q] = *(const float4*)(w0 + q * 4);
        wb1[q] = *(const float4*)(w1 + q * 4);
      }
      // prefetch W two K-steps ahead into cache (global_prefetch_b8)
      const int kp = (k0 + 2 * BK < N_IN) ? (k0 + 2 * BK) : 0;
      __builtin_prefetch(wrow0 + (size_t)kp * N_OUT, 0, 1);
    }

    // ---- fragments: two ds_load_b128 each ----
    FragU af[4];
#pragma unroll
    for (int mt = 0; mt < 4; ++mt) {
      const unsigned short* ap = &sA[(wm * 64 + mt * 16 + ar) * LDA + ak];
      af[mt].u[0] = *(const uint4*)(ap);        // K 0..7   (or 8..15)
      af[mt].u[1] = *(const uint4*)(ap + 16);   // K 16..23 (or 24..31)
    }
    FragU bfr[4];
#pragma unroll
    for (int nt = 0; nt < 4; ++nt) {
      const unsigned short* bp = &sB[(wn * 64 + nt * 16 + ar) * LDB + bk];
      bfr[nt].u[0] = *(const uint4*)(bp);       // K 0..7   (or 8..15)
      bfr[nt].u[1] = *(const uint4*)(bp + 16);  // K 16..23 (or 24..31)
    }

#pragma unroll
    for (int mt = 0; mt < 4; ++mt)
#pragma unroll
      for (int nt = 0; nt < 4; ++nt)
        acc[mt][nt] = __builtin_amdgcn_wmma_f32_16x16x32_bf16(
            false, af[mt].v, false, bfr[nt].v, (short)0, acc[mt][nt],
            false, false);
    __syncthreads();
  }

  // ---- epilogue: D layout -> lane(0..15)=N, VGPR r -> M = r + 8*(lane>=16)
  const int cn  = lane & 15;
  const int cm8 = (lane >> 4) * 8;
#pragma unroll
  for (int nt = 0; nt < 4; ++nt) {
    const int col = bn + wn * 64 + nt * 16 + cn;
    const float bia = bias[col];
#pragma unroll
    for (int mt = 0; mt < 4; ++mt) {
      const int row0 = bm + wm * 64 + mt * 16 + cm8;
#pragma unroll
      for (int r = 0; r < 8; ++r)
        out[(size_t)(row0 + r) * N_OUT + col] = acc[mt][nt][r] + bia;
    }
  }
}

// ---------------------------------------------------------------------------
extern "C" void kernel_launch(void* const* d_in, const int* in_sizes, int n_in,
                              void* d_out, int out_size, void* d_ws, size_t ws_size,
                              hipStream_t stream) {
  (void)in_sizes; (void)n_in; (void)out_size; (void)ws_size;
  const float* x    = (const float*)d_in[0];
  const float* w    = (const float*)d_in[1];
  const float* bias = (const float*)d_in[2];
  const float* vs   = (const float*)d_in[3];
  const int*   tm   = (const int*)d_in[4];
  float* out = (float*)d_out;

  float* vn = (float*)d_ws;     // 4096 f32
  float* xv = vn + N_IN;        // 8192 f32   (48 KB total scratch)

  k_norm<<<1, 256, 0, stream>>>(vs, tm, vn);
  k_xv<<<BATCH / 8, 256, 0, stream>>>(x, vn, xv);

  dim3 grid(N_OUT / BN, BATCH / BM);   // (16, 64)
  k_gemm<<<grid, 256, 0, stream>>>(x, w, bias, vn, xv, out);
}
